// WSLiftConv2d_37151467110912
// MI455X (gfx1250) — compile-verified, matrix operand order checked by reading, not run on hardware
//
#include <hip/hip_runtime.h>
#include <hip/hip_bf16.h>
#include <math.h>

typedef __attribute__((ext_vector_type(16))) _Float16 v16h;
typedef __attribute__((ext_vector_type(8)))  _Float16 v8h;
typedef __attribute__((ext_vector_type(8)))  float    v8f;

#define W_IMG 224
#define H_IMG 224
#define HW    (224 * 224)
#define OCG   512           // 64 oc * 8 g
#define KREAL 75            // ic(3) * 25 kernel taps
#define KPAD  96            // padded to 3 * 32 for f16 WMMA K=32

// ---------------------------------------------------------------------------
// Phase 1: log-space Sinkhorn, 20 iterations, one wave per group g (8 groups)
// ---------------------------------------------------------------------------
__global__ void sinkhorn_kernel(const float* __restrict__ reps,
                                float* __restrict__ rep_out) {
  __shared__ float la[625];
  const int g = blockIdx.x;
  const int t = threadIdx.x;                      // 32 threads
  for (int i = t; i < 625; i += 32) la[i] = reps[g * 625 + i];
  __syncthreads();
  for (int it = 0; it < 20; ++it) {
    if (t < 25) {                                 // row logsumexp (axis -1)
      float mx = -__builtin_inff();
      for (int m = 0; m < 25; ++m) mx = fmaxf(mx, la[t * 25 + m]);
      float s = 0.f;
      for (int m = 0; m < 25; ++m) s += expf(la[t * 25 + m] - mx);
      float lse = mx + logf(s);
      for (int m = 0; m < 25; ++m) la[t * 25 + m] -= lse;
    }
    __syncthreads();
    if (t < 25) {                                 // column logsumexp (axis -2)
      float mx = -__builtin_inff();
      for (int n = 0; n < 25; ++n) mx = fmaxf(mx, la[n * 25 + t]);
      float s = 0.f;
      for (int n = 0; n < 25; ++n) s += expf(la[n * 25 + t] - mx);
      float lse = mx + logf(s);
      for (int n = 0; n < 25; ++n) la[n * 25 + t] -= lse;
    }
    __syncthreads();
  }
  for (int i = t; i < 625; i += 32) rep_out[g * 625 + i] = expf(la[i]);
}

// ---------------------------------------------------------------------------
// Phase 2: filt[c = oc*8+g][k = ic*25+n] = sum_m rep[g,n,m] * weight[oc,ic,m]
// stored f16, K padded 75 -> 96 with zeros.
// ---------------------------------------------------------------------------
__global__ void synth_kernel(const float* __restrict__ rep,
                             const float* __restrict__ weight,
                             _Float16* __restrict__ filt) {
  int idx = blockIdx.x * 256 + threadIdx.x;       // over 512*96
  if (idx >= OCG * KPAD) return;
  int ocg = idx / KPAD;
  int r   = idx - ocg * KPAD;
  _Float16 v = (_Float16)0.f;
  if (r < KREAL) {
    int oc = ocg >> 3, g = ocg & 7;
    int ic = r / 25, n = r - ic * 25;
    const float* rp = rep + (g * 25 + n) * 25;
    const float* wp = weight + (oc * 3 + ic) * 25;
    float s = 0.f;
    for (int m = 0; m < 25; ++m) s += rp[m] * wp[m];
    v = (_Float16)s;
  }
  filt[idx] = v;
}

// ---------------------------------------------------------------------------
// Phase 3: implicit-GEMM conv via v_wmma_f32_16x16x32_f16.
// Workgroup = 256 threads (8 waves) owns a 32-pixel row tile (b, y, x0..x0+31)
// = two 16-pixel N-tiles. im2col B tiles (K=96 x N=16 each) are staged in LDS
// pre-swizzled into the exact per-lane WMMA B register layout
// (lane%16 = N, lane/16 = K-half; element j<8 -> K = half*8+j,
//  j>=8 -> K = 16+half*8+(j-8), within each kt*32 K-chunk).
// Wave w covers channels [w*64, w*64+64): 4 A-tiles (M = channels) x 3 K steps,
// each filter A-tile reused against both pixel B-tiles (amortizes A loads).
// M=channels / N=pixels makes output stores coalesced (16 consecutive pixels
// per half-wave per global_store_b32).
// ---------------------------------------------------------------------------
__global__ __launch_bounds__(256)
void conv_wmma_kernel(const float* __restrict__ x,
                      const _Float16* __restrict__ filt,
                      const float* __restrict__ bias,
                      float* __restrict__ out) {
  // layout: ldsB[((kt*2 + ht)*32 + lane)*16 + j], kt<3, ht<2 -> 3072 f16 = 6 KB
  __shared__ alignas(32) _Float16 ldsB[3 * 2 * 32 * 16];

  const int tile = blockIdx.x;                  // 8*224*7 = 12544 tiles
  const int xt   = tile % 7;
  const int x0   = xt * 32;
  const int rem  = tile / 7;
  const int y    = rem % H_IMG;
  const int b    = rem / H_IMG;
  const int tid  = threadIdx.x;

  // ---- cooperative im2col into swizzled LDS layout (12 elements / thread)
  #pragma unroll
  for (int i = 0; i < 12; ++i) {
    int flat = tid + i * 256;                   // 0..3071
    int j    = flat & 15;
    int lane = (flat >> 4) & 31;
    int ht   = (flat >> 9) & 1;
    int kt   = flat >> 10;                      // 0..2
    int n    = ht * 16 + (lane & 15);           // pixel within 32-wide tile
    int half = lane >> 4;
    int K = kt * 32 + ((j < 8) ? (half * 8 + j) : (16 + half * 8 + (j - 8)));
    _Float16 v = (_Float16)0.f;
    if (K < KREAL) {
      int ic = K / 25, r = K - ic * 25;
      int ky = r / 5,  kx = r - ky * 5;
      int xx = x0 + n + kx - 2;
      int yy = y + ky - 2;
      if (xx >= 0 && xx < W_IMG && yy >= 0 && yy < H_IMG)
        v = (_Float16)x[((size_t)(b * 3 + ic) * H_IMG + yy) * W_IMG + xx];
    }
    ldsB[flat] = v;
  }
  __syncthreads();

  const int lane = tid & 31;
  const int wave = tid >> 5;
  const int half = lane >> 4;
  const int n    = lane & 15;
  const int ch0  = wave * 64;

  v8f acc[2][4];
  #pragma unroll
  for (int ht = 0; ht < 2; ++ht)
    #pragma unroll
    for (int ct = 0; ct < 4; ++ct) acc[ht][ct] = (v8f){};

  #pragma unroll
  for (int kt = 0; kt < 3; ++kt) {
    v16h bm0 = *(const v16h*)(ldsB + ((kt * 2 + 0) * 32 + lane) * 16);
    v16h bm1 = *(const v16h*)(ldsB + ((kt * 2 + 1) * 32 + lane) * 16);
    #pragma unroll
    for (int ct = 0; ct < 4; ++ct) {
      int ch = ch0 + ct * 16 + (lane & 15);                   // A row = channel
      const _Float16* fr = filt + (size_t)ch * KPAD + kt * 32 + half * 8;
      v8h lo = *(const v8h*)(fr);        // K = kt*32 + half*8 + 0..7
      v8h hi = *(const v8h*)(fr + 16);   // K = kt*32 + 16 + half*8 + 0..7
      v16h am;
      #pragma unroll
      for (int e = 0; e < 8; ++e) { am[e] = lo[e]; am[e + 8] = hi[e]; }
      acc[0][ct] = __builtin_amdgcn_wmma_f32_16x16x32_f16(
          false, am, false, bm0, (short)0, acc[0][ct], false, false);
      acc[1][ct] = __builtin_amdgcn_wmma_f32_16x16x32_f16(
          false, am, false, bm1, (short)0, acc[1][ct], false, false);
    }
  }

  // ---- store: lane holds pixel n (of its ht half), VGPR v holds channel
  // chBase+v -> 16 consecutive f32 pixels per half-wave per store.
  size_t outBase = ((size_t)b * OCG) * HW + (size_t)y * W_IMG + x0 + n;
  #pragma unroll
  for (int ct = 0; ct < 4; ++ct) {
    float bv = bias[(ch0 >> 3) + ct * 2 + half];   // oc = ch/8, same for v=0..7
    int chBase = ch0 + ct * 16 + 8 * half;
    #pragma unroll
    for (int v = 0; v < 8; ++v) {
      size_t o = outBase + (size_t)(chBase + v) * HW;
      out[o]      = acc[0][ct][v] + bv;
      out[o + 16] = acc[1][ct][v] + bv;
    }
  }
}

// ---------------------------------------------------------------------------
extern "C" void kernel_launch(void* const* d_in, const int* in_sizes, int n_in,
                              void* d_out, int out_size, void* d_ws, size_t ws_size,
                              hipStream_t stream) {
  const float* x      = (const float*)d_in[0];   // [8,3,224,224]
  const float* weight = (const float*)d_in[1];   // [64,3,25]
  const float* reps   = (const float*)d_in[2];   // [8,25,25]
  const float* bias   = (const float*)d_in[3];   // [64]
  float* out = (float*)d_out;                    // [8,512,224,224]

  // workspace layout
  float*    rep_ws  = (float*)d_ws;                          // 8*625*4 = 20000 B
  _Float16* filt_ws = (_Float16*)((char*)d_ws + 20480);      // 512*96*2 = 98304 B

  sinkhorn_kernel<<<8, 32, 0, stream>>>(reps, rep_ws);

  synth_kernel<<<(OCG * KPAD + 255) / 256, 256, 0, stream>>>(rep_ws, weight, filt_ws);

  const int tiles = 8 * H_IMG * (W_IMG / 32);   // 12544
  conv_wmma_kernel<<<tiles, 256, 0, stream>>>(x, filt_ws, bias, out);
}
